// MoeLayer_51582557225405
// MI455X (gfx1250) — matrix operand order repaired and checked by reading, not compiled, not run on hardware
//
#include <hip/hip_runtime.h>
#include <hip/hip_bf16.h>
#include <hip/hip_fp16.h>

typedef _Float16 v16h __attribute__((ext_vector_type(16)));
typedef _Float16 v8h  __attribute__((ext_vector_type(8)));
typedef _Float16 v4h  __attribute__((ext_vector_type(4)));
typedef float    v8f  __attribute__((ext_vector_type(8)));

// async-to-LDS builtin operand types: v4i pointers in AS(1)/AS(3)
typedef int i32x4 __attribute__((vector_size(16)));
typedef __attribute__((address_space(1))) i32x4 glob_i32x4;
typedef __attribute__((address_space(3))) i32x4 lds_i32x4;

#define B_TOK 16384
#define D_IN  256
#define D_HID 512
#define D_OUT 256
#define NE    8
#define TM    64
#define TILES_PER_E (B_TOK / TM)   // 256

__device__ __forceinline__ void wait_async0() {
#if __has_builtin(__builtin_amdgcn_s_wait_asynccnt)
  __builtin_amdgcn_s_wait_asynccnt(0);
#else
  asm volatile("s_wait_asynccnt 0x0" ::: "memory");
#endif
}

// ---------------------------------------------------------------------------
// prep: zero d_out (B*D_OUT = 4,194,304) and expert counts
// ---------------------------------------------------------------------------
__global__ __launch_bounds__(256) void prep_kernel(float* __restrict__ out,
                                                   int* __restrict__ counts) {
  int idx = blockIdx.x * 256 + threadIdx.x;
  out[idx] = 0.0f;
  if (idx < NE) counts[idx] = 0;
}

// ---------------------------------------------------------------------------
// wconv: W1/W2 f32 -> f16 in WMMA B-fragment layout:
//   frag(e, ks, nt, lane)[h] = W[e][ks*32 + 16*(lane>>4) + h][nt*16 + (lane&15)]
// ---------------------------------------------------------------------------
__global__ __launch_bounds__(256) void wconv_kernel(const float* __restrict__ W1,
                                                    const float* __restrict__ W2,
                                                    _Float16* __restrict__ w1f,
                                                    _Float16* __restrict__ w2f) {
  int gid = blockIdx.x * 256 + threadIdx.x;
  if (gid < 65536) {  // W1 [e][256][512]
    int lane = gid & 31, nt = (gid >> 5) & 31, ks = (gid >> 10) & 7, e = gid >> 13;
    int n = nt * 16 + (lane & 15);
    int kbase = ks * 32 + 16 * (lane >> 4);
    _Float16* dst = w1f + (size_t)gid * 16;
    const float* src = W1 + ((size_t)e * D_IN + kbase) * D_HID + n;
#pragma unroll
    for (int h = 0; h < 16; ++h) dst[h] = (_Float16)src[h * D_HID];
  } else {            // W2 [e][512][256]
    int g = gid - 65536;
    int lane = g & 31, nt = (g >> 5) & 15, ks = (g >> 9) & 15, e = g >> 13;
    int n = nt * 16 + (lane & 15);
    int kbase = ks * 32 + 16 * (lane >> 4);
    _Float16* dst = w2f + (size_t)g * 16;
    const float* src = W2 + ((size_t)e * D_HID + kbase) * D_OUT + n;
#pragma unroll
    for (int h = 0; h < 16; ++h) dst[h] = (_Float16)src[h * D_OUT];
  }
}

// ---------------------------------------------------------------------------
// gate: one block = 32 tokens. Stage x tile in LDS (coalesced), fuse x->f16
// conversion, compute 8 logits per token, strict top-2 (ties -> lowest index,
// like lax.top_k), 2-way softmax, bucket append.
// ---------------------------------------------------------------------------
__global__ __launch_bounds__(256) void gate_kernel(const float* __restrict__ x,
                                                   const float* __restrict__ Wg,
                                                   _Float16* __restrict__ xh,
                                                   int* __restrict__ counts,
                                                   int* __restrict__ btok,
                                                   float* __restrict__ bw) {
  __shared__ __align__(16) float xs[32][D_IN];   // 32 KB
  __shared__ float wg_s[D_IN * NE];              // 8 KB
  __shared__ float lg[32][NE];
  int tid = threadIdx.x;
  int t0  = blockIdx.x * 32;

  for (int i = tid; i < D_IN * NE; i += 256) wg_s[i] = Wg[i];

  // stage 32 x-rows (8192 floats) as float4, convert to f16 on the way out
  for (int i = tid; i < 32 * (D_IN / 4); i += 256) {   // 2048 float4 chunks
    int r = i >> 6, c = i & 63;
    float4 v = ((const float4*)(x + (size_t)(t0 + r) * D_IN))[c];
    ((float4*)&xs[r][0])[c] = v;
    v4h h;
    h[0] = (_Float16)v.x; h[1] = (_Float16)v.y;
    h[2] = (_Float16)v.z; h[3] = (_Float16)v.w;
    *(v4h*)(xh + (size_t)(t0 + r) * D_IN + c * 4) = h;
  }
  __syncthreads();

  // logits: thread = (token, expert)
  int tok = tid >> 3, e = tid & 7;
  float acc = 0.0f;
  for (int d = 0; d < D_IN; ++d) acc += xs[tok][d] * wg_s[d * NE + e];
  lg[tok][e] = acc;
  __syncthreads();

  if (tid < 32) {
    int b = t0 + tid;
    float l[NE];
#pragma unroll
    for (int q = 0; q < NE; ++q) l[q] = lg[tid][q];
    int e0 = 0; float l0 = l[0];
#pragma unroll
    for (int q = 1; q < NE; ++q) if (l[q] > l0) { l0 = l[q]; e0 = q; }
    int e1 = -1; float l1 = -3.0e38f;
#pragma unroll
    for (int q = 0; q < NE; ++q) if (q != e0 && l[q] > l1) { l1 = l[q]; e1 = q; }
    float z  = __expf(l1 - l0);
    float iv = 1.0f / (1.0f + z);
    int p0 = atomicAdd(&counts[e0], 1);
    btok[e0 * B_TOK + p0] = b; bw[e0 * B_TOK + p0] = iv;
    int p1 = atomicAdd(&counts[e1], 1);
    btok[e1 * B_TOK + p1] = b; bw[e1 * B_TOK + p1] = z * iv;
  }
}

// ---------------------------------------------------------------------------
// moe_gemm: one block = TM(=64) bucket rows of one expert.
//   GEMM1: h = relu(gathered_x @ W1[e] + b1[e])   (f16 WMMA, f32 acc)
//   GEMM2: y = h @ W2[e] + b2[e]
//   epilogue: out[token] += gate_w * y  (2 commutative f32 atomics per elem)
// Gather uses GLOBAL_LOAD_ASYNC_TO_LDS when available (ASYNCcnt path).
// ---------------------------------------------------------------------------
__global__ __launch_bounds__(256) void moe_gemm_kernel(
    const _Float16* __restrict__ xh,  const _Float16* __restrict__ w1f,
    const _Float16* __restrict__ w2f, const float* __restrict__ b1,
    const float* __restrict__ b2,     const int* __restrict__ counts,
    const int* __restrict__ btok,     const float* __restrict__ bw,
    float* __restrict__ out) {
  int e    = blockIdx.x >> 8;
  int tile = blockIdx.x & 255;
  int cnt  = counts[e];
  int row0 = tile * TM;
  if (row0 >= cnt) return;                 // block-uniform: EXEC all-ones below
  int rows = min(TM, cnt - row0);

  __shared__ __align__(16) _Float16 As[TM][D_IN];    // 32 KB
  __shared__ __align__(16) _Float16 Hs[TM][D_HID];   // 64 KB
  __shared__ int   tok_s[TM];
  __shared__ float w_s[TM];

  int tid = threadIdx.x;
  if (tid < TM) {
    int t = 0; float wt = 0.0f;            // pad rows -> token 0 (valid data,
    if (tid < rows) {                      // masked out in the epilogue)
      t  = btok[e * B_TOK + row0 + tid];
      wt = bw[e * B_TOK + row0 + tid];
    }
    tok_s[tid] = t; w_s[tid] = wt;
  }
  __syncthreads();

  // gather TM rows x 256 halves as 16B chunks
#if __has_builtin(__builtin_amdgcn_global_load_async_to_lds_b128)
  for (int i = tid; i < TM * (D_IN / 8); i += 256) {
    int r = i >> 5, c = i & 31;
    glob_i32x4* g = (glob_i32x4*)(xh + (size_t)tok_s[r] * D_IN + c * 8);
    lds_i32x4*  l = (lds_i32x4*)&As[r][c * 8];
    __builtin_amdgcn_global_load_async_to_lds_b128(g, l, 0, 0);
  }
  wait_async0();
#else
  for (int i = tid; i < TM * (D_IN / 8); i += 256) {
    int r = i >> 5, c = i & 31;
    uint4 v = ((const uint4*)(xh + (size_t)tok_s[r] * D_IN))[c];
    ((uint4*)&As[r][0])[c] = v;
  }
#endif
  __syncthreads();

  int wave = tid >> 5, ln = tid & 31, hi = ln >> 4, lm = ln & 15;

  // ---------------- GEMM1: [64 x 256] x [256 x 512] -> Hs ----------------
  for (int ntb = 0; ntb < 4; ++ntb) {
    int nt = wave + ntb * 8;                 // n-tile 0..31
    v8f acc[4] = {};
    const v16h* bp =
        (const v16h*)(w1f + (((size_t)e * 256 + nt) * 32 + ln) * 16);
#pragma unroll
    for (int ks = 0; ks < 8; ++ks) {
      v16h bf = bp[(size_t)ks * 1024];       // ks stride = 32*32*16 halves
      int k0 = ks * 32 + 16 * hi;            // contiguous k order, matches B frags
#pragma unroll
      for (int mt = 0; mt < 4; ++mt) {
        v16h a;
        ((v8h*)&a)[0] = *(const v8h*)&As[mt * 16 + lm][k0];
        ((v8h*)&a)[1] = *(const v8h*)&As[mt * 16 + lm][k0 + 8];
        acc[mt] = __builtin_amdgcn_wmma_f32_16x16x32_f16(
            false, a, false, bf, (short)0, acc[mt], false, false);
      }
    }
    int col = nt * 16 + lm;
    float bias = b1[e * D_HID + col];
#pragma unroll
    for (int mt = 0; mt < 4; ++mt)
#pragma unroll
      for (int v = 0; v < 8; ++v) {          // C layout: M = v + 8*hi
        float hv = acc[mt][v] + bias;
        hv = hv > 0.0f ? hv : 0.0f;
        Hs[mt * 16 + v + 8 * hi][col] = (_Float16)hv;
      }
  }
  __syncthreads();

  // ---------------- GEMM2: [64 x 512] x [512 x 256] -> out ----------------
  for (int ntb = 0; ntb < 2; ++ntb) {
    int nt = wave + ntb * 8;                 // n-tile 0..15
    v8f acc[4] = {};
    const v16h* bp =
        (const v16h*)(w2f + (((size_t)e * 256 + nt) * 32 + ln) * 16);
#pragma unroll
    for (int ks = 0; ks < 16; ++ks) {
      v16h bf = bp[(size_t)ks * 512];        // ks stride = 16*32*16 halves
      int k0 = ks * 32 + 16 * hi;
#pragma unroll
      for (int mt = 0; mt < 4; ++mt) {
        v16h a;
        ((v8h*)&a)[0] = *(const v8h*)&Hs[mt * 16 + lm][k0];
        ((v8h*)&a)[1] = *(const v8h*)&Hs[mt * 16 + lm][k0 + 8];
        acc[mt] = __builtin_amdgcn_wmma_f32_16x16x32_f16(
            false, a, false, bf, (short)0, acc[mt], false, false);
      }
    }
    int col = nt * 16 + lm;
    float bias = b2[e * D_OUT + col];
#pragma unroll
    for (int mt = 0; mt < 4; ++mt)
#pragma unroll
      for (int v = 0; v < 8; ++v) {
        int r = mt * 16 + v + 8 * hi;
        if (r < rows)
          atomicAdd(out + (size_t)tok_s[r] * D_OUT + col,
                    (acc[mt][v] + bias) * w_s[r]);
      }
  }
}

// ---------------------------------------------------------------------------
extern "C" void kernel_launch(void* const* d_in, const int* in_sizes, int n_in,
                              void* d_out, int out_size, void* d_ws, size_t ws_size,
                              hipStream_t stream) {
  const float* x  = (const float*)d_in[0];
  const float* Wg = (const float*)d_in[1];
  const float* W1 = (const float*)d_in[2];
  const float* b1 = (const float*)d_in[3];
  const float* W2 = (const float*)d_in[4];
  const float* b2 = (const float*)d_in[5];
  float* out = (float*)d_out;

  char* w = (char*)d_ws;
  _Float16* xh  = (_Float16*)w;  w += (size_t)B_TOK * D_IN * 2;        // 8 MB
  _Float16* w1f = (_Float16*)w;  w += (size_t)NE * D_IN * D_HID * 2;   // 2 MB
  _Float16* w2f = (_Float16*)w;  w += (size_t)NE * D_HID * D_OUT * 2;  // 2 MB
  int*   counts = (int*)w;       w += 256;
  int*   btok   = (int*)w;       w += (size_t)NE * B_TOK * 4;          // 512 KB
  float* bwts   = (float*)w;     w += (size_t)NE * B_TOK * 4;          // 512 KB

  prep_kernel<<<(B_TOK * D_OUT) / 256, 256, 0, stream>>>(out, counts);
  wconv_kernel<<<512, 256, 0, stream>>>(W1, W2, w1f, w2f);
  gate_kernel<<<B_TOK / 32, 256, 0, stream>>>(x, Wg, xh, counts, btok, bwts);
  moe_gemm_kernel<<<NE * TILES_PER_E, 256, 0, stream>>>(xh, w1f, w2f, b1, b2,
                                                        counts, btok, bwts, out);
}
